// GCNWithAggregator_Resnet_9070970929659
// MI455X (gfx1250) — matrix operand order repaired and checked by reading, not compiled, not run on hardware
//
#include <hip/hip_runtime.h>
#include <hip/hip_bf16.h>

// ---------------------------------------------------------------------------
// GCNWithAggregator_Resnet for MI455X (gfx1250)
//  - Dense GEMMs: v_wmma_f32_16x16x32_bf16, bf16 operands pre-converted in
//    global memory, tiles staged to LDS with global_load_async_to_lds_b128
//    (ASYNCcnt) using double-buffered LDS for copy/compute overlap.
//  - Graph aggregation: global_atomic_add_f32 (memory bound).
// ---------------------------------------------------------------------------

#define N_NODES  50000
#define N_GENE   40000
#define N_EDGES  800000
#define GOID_DIM 4096
#define IN_DIM   512
#define HID1     512
#define HID2     256
#define GO_HID   1024

typedef __attribute__((ext_vector_type(16))) __bf16 v16bf;
typedef __attribute__((ext_vector_type(8)))  float  v8f;

static __device__ __forceinline__ unsigned short f2bf(float f) {
    unsigned u = __float_as_uint(f);
    u += 0x7fffu + ((u >> 16) & 1u);     // round-to-nearest-even
    return (unsigned short)(u >> 16);
}
static __device__ __forceinline__ unsigned pack2(float a, float b) {
    return (unsigned)f2bf(a) | ((unsigned)f2bf(b) << 16);
}

// ---------------------------------------------------------------------------
// bf16 WMMA GEMM:  C[M,N] = act(A[M,K] @ Bt[N,K]^T + bias)
//   A  : bf16 row-major MxK
//   Bt : bf16, weights pre-TRANSPOSED to NxK so both tiles are raw 64B rows.
// Block tile 128x128, BK=32. 8 waves -> each wave 32x64 = 2x4 WMMA frags.
// Tiles staged with async-to-LDS copies, double buffered.
// ---------------------------------------------------------------------------
template <int RELU, int BIAS, int OUT_BF16>
__global__ __launch_bounds__(256)
void gemm_bf16_wmma(const unsigned short* __restrict__ A,
                    const unsigned short* __restrict__ Bt,
                    const float* __restrict__ bias,
                    void* __restrict__ Cout,
                    int M, int N, int K) {
    constexpr int BM = 128, BN = 128, BK = 32, LDT = 48; // LDS stride (shorts)
    __shared__ __align__(16) unsigned short As[2][BM * LDT];
    __shared__ __align__(16) unsigned short Bs[2][BN * LDT];

    const int tid  = threadIdx.x;
    const int lane = tid & 31;
    const int wave = tid >> 5;          // 0..7
    const int wm   = (wave >> 1) * 32;  // wave row offset within block tile
    const int wn   = (wave & 1) * 64;   // wave col offset within block tile
    const int g    = lane >> 4;         // half-wave select
    const int l15  = lane & 15;

    const int m0 = blockIdx.y * BM;
    const int n0 = blockIdx.x * BN;

    // LDS byte addresses of the two staging buffers (low 32 bits of the
    // shared aperture address == LDS offset on CDNA5).
    const unsigned asb[2] = { (unsigned)(unsigned long long)&As[0][0],
                              (unsigned)(unsigned long long)&As[1][0] };
    const unsigned bsb[2] = { (unsigned)(unsigned long long)&Bs[0][0],
                              (unsigned)(unsigned long long)&Bs[1][0] };

    // Per-thread copy assignment: 512 16B segments per tile, 2 per thread.
    // seg s: row = s>>2 (0..127), q = s&3 (16B piece within the 64B row).
    const int s0row = tid >> 2,           s0q = tid & 3;
    const int s1row = (tid + 256) >> 2,   s1q = tid & 3;   // +256 => row+64
    const unsigned dA0 = (unsigned)((s0row * LDT + s0q * 8) * 2);
    const unsigned dA1 = (unsigned)((s1row * LDT + s1q * 8) * 2);

    const long long aRow0 = (long long)(m0 + s0row) * K + s0q * 8;
    const long long aRow1 = (long long)(m0 + s1row) * K + s1q * 8;
    const long long bRow0 = (long long)(n0 + s0row) * K + s0q * 8;
    const long long bRow1 = (long long)(n0 + s1row) * K + s1q * 8;

    auto stage = [&](int buf, int k0) {
        const unsigned short* ga0 = A + aRow0 + k0;
        const unsigned short* ga1 = A + aRow1 + k0;
        const unsigned short* gb0 = Bt + bRow0 + k0;
        const unsigned short* gb1 = Bt + bRow1 + k0;
        const unsigned da0 = asb[buf] + dA0, da1 = asb[buf] + dA1;
        const unsigned db0 = bsb[buf] + dA0, db1 = bsb[buf] + dA1;
        asm volatile("global_load_async_to_lds_b128 %0, %1, off"
                     :: "v"(da0), "v"(ga0) : "memory");
        asm volatile("global_load_async_to_lds_b128 %0, %1, off"
                     :: "v"(da1), "v"(ga1) : "memory");
        asm volatile("global_load_async_to_lds_b128 %0, %1, off"
                     :: "v"(db0), "v"(gb0) : "memory");
        asm volatile("global_load_async_to_lds_b128 %0, %1, off"
                     :: "v"(db1), "v"(gb1) : "memory");
    };

    v8f acc[2][4];
#pragma unroll
    for (int mi = 0; mi < 2; ++mi)
#pragma unroll
        for (int ni = 0; ni < 4; ++ni)
            acc[mi][ni] = (v8f){0.f,0.f,0.f,0.f,0.f,0.f,0.f,0.f};

    const int KT = K / BK;
    stage(0, 0);

    for (int kt = 0; kt < KT; ++kt) {
        asm volatile("s_wait_asynccnt 0x0" ::: "memory"); // my copies done
        __syncthreads();                                  // everyone's done
        const unsigned short* as = As[kt & 1];
        const unsigned short* bs = Bs[kt & 1];

        // ---- fragments (layouts per CDNA5 ISA 7.12.2) ----
        v16bf afr[2], bfr[4];
#pragma unroll
        for (int mi = 0; mi < 2; ++mi) {
            const int m = wm + mi * 16 + l15;
            union { uint4 u[2]; v16bf v; } t;
            t.u[0] = *reinterpret_cast<const uint4*>(&as[m * LDT + g * 8]);
            t.u[1] = *reinterpret_cast<const uint4*>(&as[m * LDT + 16 + g * 8]);
            afr[mi] = t.v;
        }
#pragma unroll
        for (int ni = 0; ni < 4; ++ni) {
            const int n = wn + ni * 16 + l15;
            union { uint4 u[2]; v16bf v; } t;
            t.u[0] = *reinterpret_cast<const uint4*>(&bs[n * LDT + g * 8]);
            t.u[1] = *reinterpret_cast<const uint4*>(&bs[n * LDT + 16 + g * 8]);
            bfr[ni] = t.v;
        }

        // prefetch next tile into the other buffer while we do the math
        if (kt + 1 < KT) stage((kt + 1) & 1, (kt + 1) * BK);

#pragma unroll
        for (int mi = 0; mi < 2; ++mi)
#pragma unroll
            for (int ni = 0; ni < 4; ++ni)
                acc[mi][ni] = __builtin_amdgcn_wmma_f32_16x16x32_bf16(
                    false, afr[mi], false, bfr[ni],
                    (short)0, acc[mi][ni], false, false);
    }

    // ---- epilogue: bias / relu / store (f32 or bf16) ----
#pragma unroll
    for (int ni = 0; ni < 4; ++ni) {
        const int col = n0 + wn + ni * 16 + l15;
        const float bv = BIAS ? bias[col] : 0.0f;
#pragma unroll
        for (int mi = 0; mi < 2; ++mi) {
#pragma unroll
            for (int r = 0; r < 8; ++r) {
                const int row = m0 + wm + mi * 16 + g * 8 + r;
                if (row < M) {
                    float v = acc[mi][ni][r] + bv;
                    if (RELU) v = fmaxf(v, 0.0f);
                    if (OUT_BF16)
                        ((unsigned short*)Cout)[(long long)row * N + col] = f2bf(v);
                    else
                        ((float*)Cout)[(long long)row * N + col] = v;
                }
            }
        }
    }
}

// ---------------------------------------------------------------------------
// Conversion kernels (run once per call, tiny vs total traffic)
// ---------------------------------------------------------------------------
// Wt[n*K + k] = bf16(W[k*N + n])   (transpose + convert)
__global__ void k_cvt_wt(const float* __restrict__ W, unsigned short* __restrict__ Wt,
                         int K, int N) {
    const int total = K * N;
    for (int i = blockIdx.x * blockDim.x + threadIdx.x; i < total;
         i += gridDim.x * blockDim.x) {
        const int k = i / N, n = i % N;
        Wt[(long long)n * K + k] = f2bf(W[i]);
    }
}

// o[i] = bf16(a[i]) vectorized (n4 float4 groups)
__global__ void k_cvt_a(const float* __restrict__ a, unsigned short* __restrict__ o,
                        long long n4) {
    for (long long i = (long long)blockIdx.x * blockDim.x + threadIdx.x; i < n4;
         i += (long long)gridDim.x * blockDim.x) {
        const float4 v = reinterpret_cast<const float4*>(a)[i];
        reinterpret_cast<uint2*>(o)[i] = make_uint2(pack2(v.x, v.y), pack2(v.z, v.w));
    }
}

// xt_bf[:n_gene] = bf16(x[:n_gene, :512])  (x row stride 4096)
__global__ void k_copy_gene(const float* __restrict__ x, unsigned short* __restrict__ xt) {
    const int n4 = N_GENE * (IN_DIM / 4);
    for (int i = blockIdx.x * blockDim.x + threadIdx.x; i < n4;
         i += gridDim.x * blockDim.x) {
        const int row = i >> 7;          // /128
        const int c   = i & 127;
        const float4 v =
            reinterpret_cast<const float4*>(x)[(long long)row * (GOID_DIM / 4) + c];
        reinterpret_cast<uint2*>(xt)[(long long)row * 128 + c] =
            make_uint2(pack2(v.x, v.y), pack2(v.z, v.w));
    }
}

// ---------------------------------------------------------------------------
// Graph / elementwise kernels
// ---------------------------------------------------------------------------
__global__ void k_fill(float* __restrict__ p, float v, int n) {
    for (int i = blockIdx.x * blockDim.x + threadIdx.x; i < n;
         i += gridDim.x * blockDim.x) p[i] = v;
}

__global__ void k_deg(const long long* __restrict__ dst, float* __restrict__ deg) {
    int i = blockIdx.x * blockDim.x + threadIdx.x;
    if (i < N_EDGES) atomicAdd(&deg[(int)dst[i]], 1.0f);
}

__global__ void k_dinv(const float* __restrict__ deg, float* __restrict__ dinv,
                       float* __restrict__ self_norm) {
    int i = blockIdx.x * blockDim.x + threadIdx.x;
    if (i < N_NODES) {
        float d = __frsqrt_rn(deg[i]);
        dinv[i] = d;
        self_norm[i] = d * d;
    }
}

__global__ void k_edge_norm(const long long* __restrict__ src,
                            const long long* __restrict__ dst,
                            const float* __restrict__ dinv,
                            float* __restrict__ en) {
    int i = blockIdx.x * blockDim.x + threadIdx.x;
    if (i < N_EDGES) en[i] = dinv[(int)src[i]] * dinv[(int)dst[i]];
}

// pre[i,f] = self_norm[i]*hw[i,f] + bias[f]
__global__ void k_selfterm(const float* __restrict__ hw,
                           const float* __restrict__ self_norm,
                           const float* __restrict__ bias,
                           float* __restrict__ pre, int D) {
    const int d4 = D >> 2;
    const long long n4 = (long long)N_NODES * d4;
    for (long long i = (long long)blockIdx.x * blockDim.x + threadIdx.x; i < n4;
         i += (long long)gridDim.x * blockDim.x) {
        const int row = (int)(i / d4);
        const int c4  = (int)(i % d4) * 4;
        const float s = self_norm[row];
        const float4 h = reinterpret_cast<const float4*>(hw)[i];
        float4 o;
        o.x = s * h.x + bias[c4 + 0];
        o.y = s * h.y + bias[c4 + 1];
        o.z = s * h.z + bias[c4 + 2];
        o.w = s * h.w + bias[c4 + 3];
        reinterpret_cast<float4*>(pre)[i] = o;
    }
}

// pre[dst] += en[e] * hw[src]   (one block per edge, blockDim = D/4)
__global__ void k_scatter(const long long* __restrict__ src,
                          const long long* __restrict__ dst,
                          const float* __restrict__ en,
                          const float* __restrict__ hw,
                          float* __restrict__ pre, int D) {
    const int e = blockIdx.x;
    const long long s = src[e];
    const long long d = dst[e];
    const float w = en[e];
    const int f = threadIdx.x * 4;
    const float4 v = *reinterpret_cast<const float4*>(&hw[s * D + f]);
    float* o = &pre[d * D + f];
    atomicAdd(o + 0, w * v.x);
    atomicAdd(o + 1, w * v.y);
    atomicAdd(o + 2, w * v.z);
    atomicAdd(o + 3, w * v.w);
}

// h = relu(pre) + r ; output f32 or bf16
template <int OUT_BF16>
__global__ void k_combine(const float* __restrict__ pre, const float* __restrict__ r,
                          void* __restrict__ h, long long n4) {
    for (long long i = (long long)blockIdx.x * blockDim.x + threadIdx.x; i < n4;
         i += (long long)gridDim.x * blockDim.x) {
        const float4 p = reinterpret_cast<const float4*>(pre)[i];
        const float4 q = reinterpret_cast<const float4*>(r)[i];
        float4 o;
        o.x = fmaxf(p.x, 0.f) + q.x;
        o.y = fmaxf(p.y, 0.f) + q.y;
        o.z = fmaxf(p.z, 0.f) + q.z;
        o.w = fmaxf(p.w, 0.f) + q.w;
        if (OUT_BF16)
            reinterpret_cast<uint2*>(h)[i] = make_uint2(pack2(o.x, o.y), pack2(o.z, o.w));
        else
            reinterpret_cast<float4*>(h)[i] = o;
    }
}

// out[i,:] = [h2[i,:256], istj[i]] @ W_f + b_f   (W_f: 257x2)
__global__ void k_final(const float* __restrict__ h2, const float* __restrict__ istj,
                        const float* __restrict__ Wf, const float* __restrict__ bf_,
                        float* __restrict__ out) {
    const int i = blockIdx.x * blockDim.x + threadIdx.x;
    if (i >= N_GENE) return;
    float a0 = bf_[0], a1 = bf_[1];
    const float* hr = &h2[(long long)i * HID2];
#pragma unroll 4
    for (int f = 0; f < HID2; ++f) {
        const float v = hr[f];
        a0 += v * Wf[2 * f + 0];
        a1 += v * Wf[2 * f + 1];
    }
    const float t = istj[i];
    a0 += t * Wf[2 * HID2 + 0];
    a1 += t * Wf[2 * HID2 + 1];
    out[2 * i + 0] = a0;
    out[2 * i + 1] = a1;
}

// ---------------------------------------------------------------------------
static inline int cdiv(int a, int b) { return (a + b - 1) / b; }

extern "C" void kernel_launch(void* const* d_in, const int* in_sizes, int n_in,
                              void* d_out, int out_size, void* d_ws, size_t ws_size,
                              hipStream_t stream) {
    const float*     x     = (const float*)d_in[0];
    const long long* ei    = (const long long*)d_in[1];   // int64 (2, N_EDGES)
    const float*     istj  = (const float*)d_in[2];
    const float*     W_go1 = (const float*)d_in[4];
    const float*     b_go1 = (const float*)d_in[5];
    const float*     W_go2 = (const float*)d_in[6];
    const float*     b_go2 = (const float*)d_in[7];
    const float*     W_c1  = (const float*)d_in[8];
    const float*     b_c1  = (const float*)d_in[9];
    const float*     W_c2  = (const float*)d_in[10];
    const float*     b_c2  = (const float*)d_in[11];
    const float*     W_r1  = (const float*)d_in[12];
    const float*     b_r1  = (const float*)d_in[13];
    const float*     W_r2  = (const float*)d_in[14];
    const float*     b_r2  = (const float*)d_in[15];
    const float*     W_f   = (const float*)d_in[16];
    const float*     b_f   = (const float*)d_in[17];
    float*           out   = (float*)d_out;

    const long long* src = ei;
    const long long* dst = ei + N_EDGES;
    const int M_GO = N_NODES - N_GENE;   // 10000

    // ---- workspace layout ----
    float* ws        = (float*)d_ws;
    float* deg       = ws;
    float* dinv      = deg + N_NODES;
    float* self_norm = dinv + N_NODES;
    float* en        = self_norm + N_NODES;

    unsigned short* us = (unsigned short*)(en + N_EDGES);
    unsigned short* Wgo1t  = us;  us += (size_t)GO_HID * GOID_DIM;
    unsigned short* Wgo2t  = us;  us += (size_t)IN_DIM * GO_HID;
    unsigned short* Wc1t   = us;  us += (size_t)HID1 * IN_DIM;
    unsigned short* Wr1t   = us;  us += (size_t)HID1 * IN_DIM;
    unsigned short* Wc2t   = us;  us += (size_t)HID2 * HID1;
    unsigned short* Wr2t   = us;  us += (size_t)HID2 * HID1;
    unsigned short* xgo_bf = us;  us += (size_t)M_GO * GOID_DIM;   // M-tail reads spill
    unsigned short* go1_bf = us;  us += (size_t)M_GO * GO_HID;     //   into the next
    unsigned short* xt_bf  = us;  us += (size_t)N_NODES * IN_DIM;  //   buffer (safe)
    unsigned short* h1_bf  = us;  us += (size_t)N_NODES * HID1;

    float* hw  = (float*)us;                         // 50000x512 f32 scratch
    float* pre = hw  + (size_t)N_NODES * HID1;
    float* rr  = pre + (size_t)N_NODES * HID1;
    float* h2  = hw;   // alias: hw2 dead once combine2 runs

    // ---- GCN normalization ----
    k_fill<<<cdiv(N_NODES, 256), 256, 0, stream>>>(deg, 1.0f, N_NODES);
    k_deg<<<cdiv(N_EDGES, 256), 256, 0, stream>>>(dst, deg);
    k_dinv<<<cdiv(N_NODES, 256), 256, 0, stream>>>(deg, dinv, self_norm);
    k_edge_norm<<<cdiv(N_EDGES, 256), 256, 0, stream>>>(src, dst, dinv, en);

    // ---- one-time bf16 conversions (weights transposed to NxK) ----
    k_cvt_wt<<<4096, 256, 0, stream>>>(W_go1, Wgo1t, GOID_DIM, GO_HID);
    k_cvt_wt<<<2048, 256, 0, stream>>>(W_go2, Wgo2t, GO_HID, IN_DIM);
    k_cvt_wt<<<1024, 256, 0, stream>>>(W_c1,  Wc1t,  IN_DIM, HID1);
    k_cvt_wt<<<1024, 256, 0, stream>>>(W_r1,  Wr1t,  IN_DIM, HID1);
    k_cvt_wt<<<512,  256, 0, stream>>>(W_c2,  Wc2t,  HID1, HID2);
    k_cvt_wt<<<512,  256, 0, stream>>>(W_r2,  Wr2t,  HID1, HID2);
    k_cvt_a<<<4096, 256, 0, stream>>>(x + (size_t)N_GENE * GOID_DIM, xgo_bf,
                                      (long long)M_GO * GOID_DIM / 4);

    // ---- GOID MLP ----
    gemm_bf16_wmma<1, 1, 1><<<dim3(GO_HID / 128, cdiv(M_GO, 128)), 256, 0, stream>>>(
        xgo_bf, Wgo1t, b_go1, go1_bf, M_GO, GO_HID, GOID_DIM);
    gemm_bf16_wmma<1, 1, 1><<<dim3(IN_DIM / 128, cdiv(M_GO, 128)), 256, 0, stream>>>(
        go1_bf, Wgo2t, b_go2, xt_bf + (size_t)N_GENE * IN_DIM, M_GO, IN_DIM, GO_HID);
    k_copy_gene<<<2048, 256, 0, stream>>>(x, xt_bf);

    // ---- Layer 1: h1 = relu(gcnconv(xt, W_c1)) + xt @ W_r1 + b_r1 ----
    gemm_bf16_wmma<0, 0, 0><<<dim3(HID1 / 128, cdiv(N_NODES, 128)), 256, 0, stream>>>(
        xt_bf, Wc1t, nullptr, hw, N_NODES, HID1, IN_DIM);
    k_selfterm<<<4096, 256, 0, stream>>>(hw, self_norm, b_c1, pre, HID1);
    k_scatter<<<N_EDGES, HID1 / 4, 0, stream>>>(src, dst, en, hw, pre, HID1);
    gemm_bf16_wmma<0, 1, 0><<<dim3(HID1 / 128, cdiv(N_NODES, 128)), 256, 0, stream>>>(
        xt_bf, Wr1t, b_r1, rr, N_NODES, HID1, IN_DIM);
    k_combine<1><<<4096, 256, 0, stream>>>(pre, rr, h1_bf, (long long)N_NODES * HID1 / 4);

    // ---- Layer 2: h2 = relu(gcnconv(h1, W_c2)) + h1 @ W_r2 + b_r2 ----
    gemm_bf16_wmma<0, 0, 0><<<dim3(HID2 / 128, cdiv(N_NODES, 128)), 256, 0, stream>>>(
        h1_bf, Wc2t, nullptr, hw, N_NODES, HID2, HID1);
    k_selfterm<<<4096, 256, 0, stream>>>(hw, self_norm, b_c2, pre, HID2);
    k_scatter<<<N_EDGES, HID2 / 4, 0, stream>>>(src, dst, en, hw, pre, HID2);
    gemm_bf16_wmma<0, 1, 0><<<dim3(HID2 / 128, cdiv(N_NODES, 128)), 256, 0, stream>>>(
        h1_bf, Wr2t, b_r2, rr, N_NODES, HID2, HID1);
    k_combine<0><<<4096, 256, 0, stream>>>(pre, rr, h2, (long long)N_NODES * HID2 / 4);

    // ---- Fusion head ----
    k_final<<<cdiv(N_GENE, 256), 256, 0, stream>>>(h2, istj, W_f, b_f, out);
}
